// Head_85323820303132
// MI455X (gfx1250) — compile-verified, matrix operand order checked by reading, not run on hardware
//
#include <hip/hip_runtime.h>
#include <hip/hip_bf16.h>
#include <math.h>

// Shapes from reference
constexpr int cB = 256;   // batch
constexpr int cT = 256;   // sequence
constexpr int cE = 384;   // embed
constexpr int cH = 64;    // head size

// ---- WMMA types (gfx1250, wave32) ----
typedef __attribute__((ext_vector_type(16))) __bf16 bf16x16;
typedef __attribute__((ext_vector_type(8)))  __bf16 bf16x8;
typedef __attribute__((ext_vector_type(8)))  float  f32x8;
typedef __attribute__((ext_vector_type(4)))  float  f32x4;

__device__ __forceinline__ f32x8 wmma_bf16f32(bf16x16 a, bf16x16 b, f32x8 c) {
  // D = A(16x32 bf16) * B(32x16 bf16) + C(16x16 f32)
  return __builtin_amdgcn_wmma_f32_16x16x32_bf16(
      /*neg_a=*/false, a, /*neg_b=*/false, b,
      /*c_mod=*/(short)0, c, /*reuse_a=*/false, /*reuse_b=*/false);
}

// Assemble a 16-halfword fragment from two contiguous 8-halfword (16B) chunks.
// Per ISA 7.12.2 (16-bit 16x32 A / 32x16 B): lane<16 -> K groups {0..7, 16..23},
// lane>=16 -> K groups {8..15, 24..31}; caller passes kbase = (lane&16)?8:0.
__device__ __forceinline__ bf16x16 frag2(const __bf16* p0, const __bf16* p1) {
  bf16x8 lo = *(const bf16x8*)p0;
  bf16x8 hi = *(const bf16x8*)p1;
  bf16x16 r;
#pragma unroll
  for (int i = 0; i < 8; ++i) { r[i] = lo[i]; r[i + 8] = hi[i]; }
  return r;
}

// ============================================================================
// Kernel 1: repack Wq/Wk/Wv fp32 [E,H] -> bf16 WMMA B-fragment layout:
//   Wf[mat*24576 + kk*2048 + c*512 + lane*16 + i] (16 contiguous halfwords/lane)
// so each wave's B-fragment is one coalesced 32B load per lane.
// ============================================================================
__global__ void pack_w_kernel(const float* __restrict__ Wq,
                              const float* __restrict__ Wk,
                              const float* __restrict__ Wv,
                              __bf16* __restrict__ Wf) {
  int id = blockIdx.x * 256 + threadIdx.x;   // grid sized exactly: 3*24576
  int mat  = id / 24576;                     // 12 kk * 4 c * 32 lanes * 16 i
  int rem  = id % 24576;
  int kk   = rem / 2048;
  int c    = (rem / 512) & 3;
  int lane = (rem / 16) & 31;
  int i    = rem & 15;
  int kl   = ((lane & 16) ? 8 : 0) + (i < 8 ? i : 16 + (i - 8));
  int k    = kk * 32 + kl;
  int n    = c * 16 + (lane & 15);
  const float* W = (mat == 0) ? Wq : (mat == 1 ? Wk : Wv);
  Wf[id] = (__bf16)W[k * cH + n];
}

// ============================================================================
// Kernel 2: fused QKV projection. WG = (batch, 64-row tile of T), 8 waves.
// x tile staged as bf16 in LDS. Each wave owns a 32x16 output tile (two
// 16-row A-fragments sharing every B-fragment -> 2 wmma per B load) and
// produces it for Q, K and V. B-fragments for kk+1 are loaded in rotating
// registers before the wmmas of kk (software pipeline, fully unrolled).
// Q,K -> bf16 row-major [b][t][h];  V -> bf16 transposed [b][h][t].
// ============================================================================
#define XSTRIDE 400   // 384 + 16 halfword pad
__global__ void qkv_proj_kernel(const float* __restrict__ x,
                                const __bf16* __restrict__ Wf,
                                __bf16* __restrict__ Qb,
                                __bf16* __restrict__ Kb,
                                __bf16* __restrict__ Vt) {
  __shared__ __align__(16) __bf16 ldsX[64 * XSTRIDE];   // 50 KB

  const int wg    = blockIdx.x;
  const int b     = wg >> 2;           // 4 row-tiles of 64 per batch
  const int tt    = wg & 3;
  const int trow0 = tt * 64;
  const int tid   = threadIdx.x;

  // ---- stage x[b, trow0:trow0+64, :] -> bf16 LDS (float4 coalesced) ----
  const float* xb = x + ((size_t)b * cT + trow0) * cE;
#pragma unroll
  for (int it = 0; it < 24; ++it) {          // 64*384/4 = 6144 float4 / 256 thr
    int v    = it * 256 + tid;
    int row  = v / 96;                        // 96 float4 per row
    int col4 = v % 96;
    f32x4 f = *(const f32x4*)(xb + row * cE + col4 * 4);
    __bf16* d = &ldsX[row * XSTRIDE + col4 * 4];
    d[0] = (__bf16)f.x; d[1] = (__bf16)f.y; d[2] = (__bf16)f.z; d[3] = (__bf16)f.w;
  }
  __syncthreads();

  const int wave  = tid >> 5;
  const int lane  = tid & 31;
  const int rp    = wave >> 2;               // 0..1 : 32-row subtile
  const int c     = wave & 3;                // 0..3 : 16-col subtile of H
  const int m     = lane & 15;
  const int half  = lane >> 4;
  const int kbase = half ? 8 : 0;

  f32x8 zero = {0,0,0,0,0,0,0,0};
  f32x8 accQ0 = zero, accQ1 = zero;
  f32x8 accK0 = zero, accK1 = zero;
  f32x8 accV0 = zero, accV1 = zero;

  const __bf16* aRow0 = &ldsX[(rp * 32 + m) * XSTRIDE];
  const __bf16* aRow1 = &ldsX[(rp * 32 + 16 + m) * XSTRIDE];
  const __bf16* wbase = Wf + ((size_t)c * 32 + lane) * 16;  // + kk*2048 + mat*24576

  // software pipeline: B fragments for kk+1 in flight while kk computes
  bf16x16 bq = *(const bf16x16*)(wbase);
  bf16x16 bk = *(const bf16x16*)(wbase + 24576);
  bf16x16 bv = *(const bf16x16*)(wbase + 49152);
#pragma unroll
  for (int kk = 0; kk < 12; ++kk) {          // E/32 K-steps
    bf16x16 a0 = frag2(aRow0 + kk * 32 + kbase, aRow0 + kk * 32 + kbase + 16);
    bf16x16 a1 = frag2(aRow1 + kk * 32 + kbase, aRow1 + kk * 32 + kbase + 16);
    bf16x16 nbq = bq, nbk = bk, nbv = bv;
    if (kk < 11) {                           // folds away under full unroll
      const __bf16* wn = wbase + (size_t)(kk + 1) * 2048;
      nbq = *(const bf16x16*)(wn);
      nbk = *(const bf16x16*)(wn + 24576);
      nbv = *(const bf16x16*)(wn + 49152);
    }
    accQ0 = wmma_bf16f32(a0, bq, accQ0);
    accQ1 = wmma_bf16f32(a1, bq, accQ1);
    accK0 = wmma_bf16f32(a0, bk, accK0);
    accK1 = wmma_bf16f32(a1, bk, accK1);
    accV0 = wmma_bf16f32(a0, bv, accV0);
    accV1 = wmma_bf16f32(a1, bv, accV1);
    bq = nbq; bk = nbk; bv = nbv;
  }

  // ---- epilogue: D layout = lane holds col n=m, VGPR p holds row p+8*half ----
  const int hcol = c * 16 + m;
#pragma unroll
  for (int rs = 0; rs < 2; ++rs) {
    const int t0 = trow0 + rp * 32 + rs * 16;
    f32x8 aq = rs ? accQ1 : accQ0;
    f32x8 ak = rs ? accK1 : accK0;
    f32x8 av = rs ? accV1 : accV0;
    size_t qk = ((size_t)b * cT + t0 + half * 8) * cH + hcol;
#pragma unroll
    for (int p = 0; p < 8; ++p) {
      Qb[qk + (size_t)p * cH] = (__bf16)aq[p];
      Kb[qk + (size_t)p * cH] = (__bf16)ak[p];
    }
    bf16x8 vp;
#pragma unroll
    for (int p = 0; p < 8; ++p) vp[p] = (__bf16)av[p];
    *(bf16x8*)(Vt + ((size_t)b * cH + hcol) * cT + t0 + half * 8) = vp;
  }
}

// ============================================================================
// Kernel 3: causal flash attention. Wave owns 16 query rows; streams key
// pairs (32 keys = one WMMA K-step). Online softmax; P transposed D->A
// layout through a per-wave LDS buffer (same-wave LDS is in-order).
// ============================================================================
__global__ void attn_kernel(const __bf16* __restrict__ Qb,
                            const __bf16* __restrict__ Kb,
                            const __bf16* __restrict__ Vt,
                            float* __restrict__ out) {
  __shared__ __align__(16) __bf16 ldsP[8][16 * 32];   // per-wave P tile

  const int tid   = threadIdx.x;
  const int wave  = tid >> 5;
  const int lane  = tid & 31;
  const int b     = blockIdx.x >> 1;
  const int qt    = (blockIdx.x & 1) * 8 + wave;      // 16 q-tiles per batch
  const int t0    = qt * 16;
  const int m     = lane & 15;
  const int half  = lane >> 4;
  const int kbase = half ? 8 : 0;

  // Q A-fragments (H=64 -> two K=32 fragments), bf16 row-major [t][h]
  const __bf16* qrow = Qb + ((size_t)b * cT + t0 + m) * cH;
  bf16x16 aq0 = frag2(qrow + kbase,      qrow + kbase + 16);
  bf16x16 aq1 = frag2(qrow + 32 + kbase, qrow + 32 + kbase + 16);

  f32x8 o0 = {0,0,0,0,0,0,0,0}, o1 = o0, o2 = o0, o3 = o0;
  float mrow[8], lrow[8];
#pragma unroll
  for (int p = 0; p < 8; ++p) { mrow[p] = -__builtin_inff(); lrow[p] = 0.f; }

  __bf16* pbuf = &ldsP[wave][0];
  const float sc = 0.125f;                            // H^-0.5
  const int npairs = (qt >> 1) + 1;                   // causal key range

  for (int jp = 0; jp < npairs; ++jp) {
    const __bf16* k0row = Kb + ((size_t)b * cT + jp * 32 + m) * cH;
    const __bf16* k1row = k0row + 16 * cH;
    const __bf16* vrow  = Vt + ((size_t)b * cH + m) * cT + jp * 32 + kbase;
    if (jp + 1 < npairs) {                            // -> global_prefetch
      __builtin_prefetch(k0row + 32 * cH, 0, 0);
      __builtin_prefetch(vrow + 32, 0, 0);
    }

    bf16x16 b00 = frag2(k0row + kbase,      k0row + kbase + 16);
    bf16x16 b01 = frag2(k0row + 32 + kbase, k0row + 32 + kbase + 16);
    bf16x16 b10 = frag2(k1row + kbase,      k1row + kbase + 16);
    bf16x16 b11 = frag2(k1row + 32 + kbase, k1row + 32 + kbase + 16);

    f32x8 z = {0,0,0,0,0,0,0,0};
    f32x8 s0 = wmma_bf16f32(aq1, b01, wmma_bf16f32(aq0, b00, z));
    f32x8 s1 = wmma_bf16f32(aq1, b11, wmma_bf16f32(aq0, b10, z));

    // scale + causal mask (arithmetic, EXEC stays all-1s) + row max
    float mx[8];
#pragma unroll
    for (int p = 0; p < 8; ++p) {
      int query = t0 + p + half * 8;
      float v0 = ((jp * 32 + m)      <= query) ? s0[p] * sc : -__builtin_inff();
      float v1 = ((jp * 32 + 16 + m) <= query) ? s1[p] * sc : -__builtin_inff();
      s0[p] = v0; s1[p] = v1;
      float v = fmaxf(v0, v1);
      v = fmaxf(v, __shfl_xor(v, 1));
      v = fmaxf(v, __shfl_xor(v, 2));
      v = fmaxf(v, __shfl_xor(v, 4));
      v = fmaxf(v, __shfl_xor(v, 8));    // stays within 16-lane half
      mx[p] = v;
    }

    float rescale[8];
#pragma unroll
    for (int p = 0; p < 8; ++p) {
      float mn = fmaxf(mrow[p], mx[p]);
      float rs = __expf(mrow[p] - mn);   // 0 on first iteration
      mrow[p] = mn; rescale[p] = rs;
      float e0 = __expf(s0[p] - mn);
      float e1 = __expf(s1[p] - mn);
      // D layout -> A layout transpose staging (bf16 in LDS)
      pbuf[(p + half * 8) * 32 + m]      = (__bf16)e0;
      pbuf[(p + half * 8) * 32 + 16 + m] = (__bf16)e1;
      float es = e0 + e1;
      es += __shfl_xor(es, 1);
      es += __shfl_xor(es, 2);
      es += __shfl_xor(es, 4);
      es += __shfl_xor(es, 8);
      lrow[p] = lrow[p] * rs + es;
    }
#pragma unroll
    for (int p = 0; p < 8; ++p) {
      o0[p] *= rescale[p]; o1[p] *= rescale[p];
      o2[p] *= rescale[p]; o3[p] *= rescale[p];
    }

    // reload P as A-fragment (same-wave LDS ops are in-order; no barrier)
    const __bf16* prow = &pbuf[m * 32];
    bf16x16 ap = frag2(prow + kbase, prow + kbase + 16);

    // O += P(16x32) * V(32x64), V transposed [h][t] -> contiguous B-fragments
    o0 = wmma_bf16f32(ap, frag2(vrow,            vrow + 16),            o0);
    o1 = wmma_bf16f32(ap, frag2(vrow + 16 * cT,  vrow + 16 * cT + 16),  o1);
    o2 = wmma_bf16f32(ap, frag2(vrow + 32 * cT,  vrow + 32 * cT + 16),  o2);
    o3 = wmma_bf16f32(ap, frag2(vrow + 48 * cT,  vrow + 48 * cT + 16),  o3);
  }

  // epilogue: out fp32 [b][t][h]
  float* orow = out + ((size_t)b * cT + t0 + half * 8) * cH + m;
#pragma unroll
  for (int p = 0; p < 8; ++p) {
    float inv = 1.0f / lrow[p];
    orow[p * cH +  0] = o0[p] * inv;
    orow[p * cH + 16] = o1[p] * inv;
    orow[p * cH + 32] = o2[p] * inv;
    orow[p * cH + 48] = o3[p] * inv;
  }
}

// ============================================================================
// Launch: pack weights -> fused QKV projection -> causal flash attention.
// Workspace: Wf (144 KB) | Qb (8 MB) | Kb (8 MB) | Vt (8 MB)  ~= 24.2 MB,
// all bf16 and fully L2-resident on MI455X (192 MB L2).
// ============================================================================
extern "C" void kernel_launch(void* const* d_in, const int* in_sizes, int n_in,
                              void* d_out, int out_size, void* d_ws, size_t ws_size,
                              hipStream_t stream) {
  const float* x  = (const float*)d_in[0];
  const float* Wq = (const float*)d_in[1];
  const float* Wk = (const float*)d_in[2];
  const float* Wv = (const float*)d_in[3];
  float* out = (float*)d_out;

  char* ws = (char*)d_ws;
  const size_t wfBytes = 3 * 24576 * sizeof(__bf16);            // 147456
  const size_t matB    = (size_t)cB * cT * cH * sizeof(__bf16); // 8 MB
  __bf16* Wf = (__bf16*)ws;
  __bf16* Qb = (__bf16*)(ws + wfBytes);
  __bf16* Kb = (__bf16*)(ws + wfBytes + matB);
  __bf16* Vt = (__bf16*)(ws + wfBytes + 2 * matB);

  pack_w_kernel  <<<(3 * 24576) / 256, 256, 0, stream>>>(Wq, Wk, Wv, Wf);
  qkv_proj_kernel<<<cB * (cT / 64),    256, 0, stream>>>(x, Wf, Qb, Kb, Vt);
  attn_kernel    <<<cB * 2,            256, 0, stream>>>(Qb, Kb, Vt, out);
}
